// HybridGraphTransformer_62912680952408
// MI455X (gfx1250) — matrix operand order repaired and checked by reading, not compiled, not run on hardware
//
#include <hip/hip_runtime.h>
#include <stdint.h>

// ---------------------------------------------------------------------------
// HybridGraphTransformer on MI455X (gfx1250, wave32, WMMA)
//   bf16 A/B WMMA (v_wmma_f32_16x16x32_bf16) for all dense GEMMs,
//   L2-resident gathers + f32 atomics for the edge-softmax segments.
// ---------------------------------------------------------------------------

typedef __attribute__((ext_vector_type(16))) __bf16 v16bf;
typedef __attribute__((ext_vector_type(8)))  float  v8f;

#define N_NODES 100000
#define HID     128
#define HEADS   4
#define DH      32
#define CD      768
#define NE      500000
#define EVN     100000
#define MPAD    100096   // multiple of 256, >= max(N_NODES, EVN)

static __device__ inline __bf16 f2bf(float f) {
  unsigned int u = __float_as_uint(f);
  unsigned int r = (u + 0x7FFFu + ((u >> 16) & 1u)) >> 16;   // RNE
  unsigned short s = (unsigned short)r;
  __bf16 b; __builtin_memcpy(&b, &s, 2); return b;
}
static __device__ inline float bf2f(__bf16 b) {
  unsigned short s; __builtin_memcpy(&s, &b, 2);
  return __uint_as_float(((unsigned int)s) << 16);
}

static __device__ inline void atomicMaxF(float* addr, float val) {
  int old = __float_as_int(*addr);
  while (__int_as_float(old) < val) {
    int assumed = old;
    old = atomicCAS((int*)addr, assumed, __float_as_int(val));
    if (old == assumed) break;
  }
}

// ---------------------------------------------------------------------------
// Pack a row-major f32 weight [Kd x Nd] into WMMA B-fragment order (bf16).
// Tile (kb, nb) of 32(K) x 16(N); lane n<16 holds B[k=0..15, n] (2/VGPR),
// lane n>=16 holds B[k=16..31, n-16]. Each lane's 16 values are contiguous.
// slot = (kb*NTtot + nb0 + nb)*512 + lane*16 + j
// ---------------------------------------------------------------------------
__global__ void pack_weight_kernel(const float* __restrict__ W, __bf16* __restrict__ dst,
                                   int Kd, int Nd, int NTtot, int nb0) {
  int t = blockIdx.x * blockDim.x + threadIdx.x;
  if (t >= Kd * Nd) return;
  int within    = t & 511;
  int tileLocal = t >> 9;
  int lane = within >> 4;
  int j    = within & 15;
  int ntLocal = Nd >> 4;
  int kb = tileLocal / ntLocal;
  int nb = tileLocal % ntLocal;
  int k   = kb * 32 + (lane >> 4) * 16 + j;
  int col = nb * 16 + (lane & 15);
  size_t slot = ((size_t)(kb * NTtot + nb0 + nb) << 9) + (size_t)within;
  dst[slot] = f2bf(W[(size_t)k * Nd + col]);
}

__global__ void pack_bias4_kernel(const float* __restrict__ bq, const float* __restrict__ bk,
                                  const float* __restrict__ bv, const float* __restrict__ bs,
                                  float* __restrict__ out) {
  int t = blockIdx.x * blockDim.x + threadIdx.x;
  if (t >= 512) return;
  const float* src = (t < 128) ? bq : (t < 256) ? bk : (t < 384) ? bv : bs;
  out[t] = src[t & 127];
}

__global__ void init_z_kernel(const float* __restrict__ emb, __bf16* __restrict__ zbf) {
  int t = blockIdx.x * blockDim.x + threadIdx.x;
  if (t >= MPAD * HID) return;
  int row = t >> 7;
  zbf[t] = (row < N_NODES) ? f2bf(emb[t]) : f2bf(0.f);
}

// ---------------------------------------------------------------------------
// Generic WMMA GEMM:  out[M x NOUT] = A[M x K] * W + bias  (opt relu / mask)
// Wave computes 32(M) x 64(N): 2 A frags x 4 B frags -> 8 WMMAs per 32-K step.
// Block = 8 waves = 256 rows. grid.x = (MPAD/256) * (NOUT/64).
// ---------------------------------------------------------------------------
__global__ __launch_bounds__(256) void gemm_wmma_kernel(
    const void* __restrict__ Avoid, int aIsF32, int aStride, int Mreal,
    const __bf16* __restrict__ Bp, const float* __restrict__ bias,
    const unsigned char* __restrict__ rowMask,
    float* __restrict__ outF, __bf16* __restrict__ outB,
    int K, int NOUT, int doRelu) {
  const int ntile64 = NOUT >> 6;
  const int mblk = blockIdx.x / ntile64;
  const int nblk = blockIdx.x % ntile64;
  const int wave = threadIdx.x >> 5;
  const int lane = threadIdx.x & 31;
  const int row0 = mblk * 256 + wave * 32;
  const int n0   = nblk * 64;
  const int NTtot = NOUT >> 4;
  const int ksteps = K >> 5;

  auto loadA = [&](int rowBase, int kBase) -> v16bf {
    int half = lane >> 4;
    int m = rowBase + (lane & 15);
    if (m >= Mreal) m = Mreal - 1;             // clamp (padded rows)
    union { v16bf v; uint4 q[2]; } u;
    if (!aIsF32) {
      const __bf16* p = (const __bf16*)Avoid + (size_t)m * aStride + kBase + half * 16;
      u.q[0] = *(const uint4*)(p);
      u.q[1] = *(const uint4*)(p + 8);
    } else {
      const float* p = (const float*)Avoid + (size_t)m * aStride + kBase + half * 16;
#pragma unroll
      for (int g = 0; g < 2; g++) {
        float4 lo = *(const float4*)(p + g * 8);
        float4 hi = *(const float4*)(p + g * 8 + 4);
        __bf16* d = ((__bf16*)&u.v) + g * 8;
        d[0] = f2bf(lo.x); d[1] = f2bf(lo.y); d[2] = f2bf(lo.z); d[3] = f2bf(lo.w);
        d[4] = f2bf(hi.x); d[5] = f2bf(hi.y); d[6] = f2bf(hi.z); d[7] = f2bf(hi.w);
      }
    }
    return u.v;
  };

  const v8f zerov = {0.f, 0.f, 0.f, 0.f, 0.f, 0.f, 0.f, 0.f};
  v8f acc[2][4];
#pragma unroll
  for (int i = 0; i < 2; i++)
#pragma unroll
    for (int j = 0; j < 4; j++) acc[i][j] = zerov;

  for (int kb = 0; kb < ksteps; ++kb) {
    const int kBase = kb << 5;
    if (kb + 1 < ksteps) {                      // prefetch next A chunk
      size_t esz = aIsF32 ? 4u : 2u;
      const char* pf = (const char*)Avoid +
          ((size_t)(row0 + (lane & 15)) * aStride + (size_t)kBase + 32) * esz;
      __builtin_prefetch(pf, 0, 3);
    }
    v16bf a0 = loadA(row0,      kBase);
    v16bf a1 = loadA(row0 + 16, kBase);
    v16bf b[4];
#pragma unroll
    for (int j = 0; j < 4; j++) {
      size_t tile = (size_t)(kb * NTtot + (n0 >> 4) + j) << 9;
      const uint4* bp = (const uint4*)(Bp + tile + lane * 16);
      union { v16bf v; uint4 q[2]; } ub;
      ub.q[0] = bp[0]; ub.q[1] = bp[1];
      b[j] = ub.v;
    }
#pragma unroll
    for (int j = 0; j < 4; j++) {
      acc[0][j] = __builtin_amdgcn_wmma_f32_16x16x32_bf16(
          false, a0, false, b[j], (short)0, acc[0][j], false, false);
      acc[1][j] = __builtin_amdgcn_wmma_f32_16x16x32_bf16(
          false, a1, false, b[j], (short)0, acc[1][j], false, false);
    }
  }

  // Epilogue: C/D layout - lane l, vgpr r -> (M = (l>>4)*8 + r, N = l&15)
#pragma unroll
  for (int i = 0; i < 2; i++) {
#pragma unroll
    for (int j = 0; j < 4; j++) {
      const int col = n0 + j * 16 + (lane & 15);
      const float bv = bias ? bias[col] : 0.f;
#pragma unroll
      for (int r = 0; r < 8; r++) {
        int row = row0 + i * 16 + (lane >> 4) * 8 + r;
        float v = acc[i][j][r] + bv;
        if (doRelu) v = fmaxf(v, 0.f);
        if (rowMask) {
          int mr = row < Mreal ? row : Mreal - 1;
          if (!rowMask[mr]) v = 0.f;
        }
        size_t oidx = (size_t)row * NOUT + col;
        if (outF) outF[oidx] = v;
        if (outB) outB[oidx] = f2bf(v);
      }
    }
  }
}

// ---------------------------------------------------------------------------
// Edge attention (TransformerConv), qkvs layout: [row*512] = q|k|v|skip
// ---------------------------------------------------------------------------
__global__ void init_attn_kernel(float* __restrict__ agg, float* __restrict__ mbuf,
                                 float* __restrict__ denom) {
  int t = blockIdx.x * blockDim.x + threadIdx.x;
  if (t < MPAD * HID) agg[t] = 0.f;
  if (t < MPAD * HEADS) { mbuf[t] = -3.0e38f; denom[t] = 0.f; }
}

__global__ void attn_scores_kernel(const int* __restrict__ ei, const float* __restrict__ qkvs,
                                   float* __restrict__ scores, float* __restrict__ mbuf) {
  int wave = threadIdx.x >> 5, lane = threadIdx.x & 31;
  int e = blockIdx.x * 8 + wave;
  if (e >= NE) return;
  int src = ei[e], dst = ei[NE + e];
  int h = lane >> 3, sub = lane & 7;
  float4 q = *(const float4*)(qkvs + (size_t)dst * 512 +       h * DH + sub * 4);
  float4 k = *(const float4*)(qkvs + (size_t)src * 512 + 128 + h * DH + sub * 4);
  float d = q.x * k.x + q.y * k.y + q.z * k.z + q.w * k.w;
  d += __shfl_xor(d, 1, 32);
  d += __shfl_xor(d, 2, 32);
  d += __shfl_xor(d, 4, 32);
  if (sub == 0) {
    float s = d * 0.17677669529663687f;         // 1/sqrt(DH)
    scores[(size_t)e * 4 + h] = s;
    atomicMaxF(&mbuf[(size_t)dst * 4 + h], s);
  }
}

__global__ void attn_accum_kernel(const int* __restrict__ ei, const float* __restrict__ qkvs,
                                  const float* __restrict__ scores, const float* __restrict__ mbuf,
                                  float* __restrict__ denom, float* __restrict__ agg) {
  int wave = threadIdx.x >> 5, lane = threadIdx.x & 31;
  int e = blockIdx.x * 8 + wave;
  if (e >= NE) return;
  int src = ei[e], dst = ei[NE + e];
  int h = lane >> 3, sub = lane & 7;
  float s  = scores[(size_t)e * 4 + h];
  float mx = mbuf[(size_t)dst * 4 + h];
  float ex = __expf(s - mx);
  if (sub == 0) atomicAdd(&denom[(size_t)dst * 4 + h], ex);
  float4 v = *(const float4*)(qkvs + (size_t)src * 512 + 256 + h * DH + sub * 4);
  float* ap = agg + (size_t)dst * HID + h * DH + sub * 4;
  atomicAdd(ap + 0, ex * v.x);
  atomicAdd(ap + 1, ex * v.y);
  atomicAdd(ap + 2, ex * v.z);
  atomicAdd(ap + 3, ex * v.w);
}

__global__ void attn_finalize_kernel(const float* __restrict__ agg, const float* __restrict__ denom,
                                     const float* __restrict__ qkvs, __bf16* __restrict__ zbf) {
  int t = blockIdx.x * blockDim.x + threadIdx.x;
  if (t >= MPAD * HID) return;
  int row = t >> 7, col = t & 127, h = col >> 5;
  float d = denom[(size_t)row * 4 + h];
  float a = agg[t];
  float s = qkvs[(size_t)row * 512 + 384 + col];          // lin_skip output
  float o = (d > 0.f ? a / d : 0.f) + s;
  zbf[t] = f2bf(fmaxf(o, 0.f));                            // relu
}

// ---------------------------------------------------------------------------
// Decoder: gather [z_u | z_v | c_u | c_v] rows (bf16), GEMM 512->128 (relu),
// then per-edge dot with W2.
// ---------------------------------------------------------------------------
__global__ void decode_gather_kernel(const int* __restrict__ edges, const __bf16* __restrict__ zbf,
                                     const __bf16* __restrict__ cbf, __bf16* __restrict__ adec) {
  int t = blockIdx.x * blockDim.x + threadIdx.x;   // one uint4 (8 bf16) per thread
  int e = t >> 6;                                  // 64 chunks per 512-wide row
  int seg = t & 63;
  if (e >= MPAD) return;
  uint4 val = {0u, 0u, 0u, 0u};
  if (e < EVN) {
    int quarter = seg >> 4;                        // 0:z[u] 1:z[v] 2:c[u] 3:c[v]
    int off = (seg & 15) * 8;
    int node = (quarter & 1) ? edges[EVN + e] : edges[e];
    const __bf16* basep = (quarter < 2) ? zbf : cbf;
    val = *(const uint4*)(basep + (size_t)node * HID + off);
  }
  *(uint4*)(adec + (size_t)e * 512 + seg * 8) = val;
}

__global__ void decode_out_kernel(const __bf16* __restrict__ hidden, const float* __restrict__ W2,
                                  const float* __restrict__ b2, float* __restrict__ out) {
  int wave = threadIdx.x >> 5, lane = threadIdx.x & 31;
  int e = blockIdx.x * 8 + wave;
  if (e >= EVN) return;
  const __bf16* hp = hidden + (size_t)e * HID + lane * 4;
  float4 w = *(const float4*)(W2 + lane * 4);
  float s = bf2f(hp[0]) * w.x + bf2f(hp[1]) * w.y + bf2f(hp[2]) * w.z + bf2f(hp[3]) * w.w;
#pragma unroll
  for (int off = 16; off; off >>= 1) s += __shfl_xor(s, off, 32);
  if (lane == 0) out[e] = s + b2[0];
}

// ---------------------------------------------------------------------------
extern "C" void kernel_launch(void* const* d_in, const int* in_sizes, int n_in,
                              void* d_out, int out_size, void* d_ws, size_t ws_size,
                              hipStream_t stream) {
  (void)in_sizes; (void)n_in; (void)out_size; (void)ws_size;
  // Input order: top-level dict insertion order; params flattened as a pytree
  // (dict keys sorted): W1, W2, Wc, b1, b2, bc, layers[ Wk,Wq,Ws,Wv,bk,bq,bs,bv ]x2
  const int*   edge_index = (const int*)d_in[0];
  const float* chemistry  = (const float*)d_in[1];
  const int*   pos_edge   = (const int*)d_in[2];
  const int*   neg_edge   = (const int*)d_in[3];
  const unsigned char* smiles_mask = (const unsigned char*)d_in[4];
  const float* node_emb   = (const float*)d_in[5];
  const float* W1 = (const float*)d_in[6];
  const float* W2 = (const float*)d_in[7];
  const float* Wc = (const float*)d_in[8];
  const float* b1 = (const float*)d_in[9];
  const float* b2 = (const float*)d_in[10];
  const float* bc = (const float*)d_in[11];

  const float* LW[2][4]; const float* LB[2][4];   // order: q,k,v,s
  for (int l = 0; l < 2; l++) {
    int base = 12 + l * 8;
    LW[l][0] = (const float*)d_in[base + 1];  // Wq
    LW[l][1] = (const float*)d_in[base + 0];  // Wk
    LW[l][2] = (const float*)d_in[base + 3];  // Wv
    LW[l][3] = (const float*)d_in[base + 2];  // Ws
    LB[l][0] = (const float*)d_in[base + 5];  // bq
    LB[l][1] = (const float*)d_in[base + 4];  // bk
    LB[l][2] = (const float*)d_in[base + 7];  // bv
    LB[l][3] = (const float*)d_in[base + 6];  // bs
  }

  // workspace carve (256B aligned slabs)
  char* ws = (char*)d_ws;
  size_t off = 0;
  auto carve = [&](size_t bytes) -> char* {
    char* p = ws + off; off += (bytes + 255) & ~(size_t)255; return p;
  };
  __bf16* zbf    = (__bf16*)carve((size_t)MPAD * HID * 2);
  __bf16* cbf    = (__bf16*)carve((size_t)MPAD * HID * 2);
  float*  qkvs   = (float*) carve((size_t)MPAD * 512 * 4);
  float*  agg    = (float*) carve((size_t)MPAD * HID * 4);
  float*  mbuf   = (float*) carve((size_t)MPAD * 4 * 4);
  float*  denom  = (float*) carve((size_t)MPAD * 4 * 4);
  float*  sc     = (float*) carve((size_t)NE * 4 * 4);
  __bf16* adec   = (__bf16*)carve((size_t)MPAD * 512 * 2);
  __bf16* hidden = (__bf16*)carve((size_t)MPAD * HID * 2);
  __bf16* packL[2];
  packL[0] = (__bf16*)carve(128 * 512 * 2);
  packL[1] = (__bf16*)carve(128 * 512 * 2);
  __bf16* packWc = (__bf16*)carve(768 * 128 * 2);
  __bf16* packW1 = (__bf16*)carve(512 * 128 * 2);
  float* bias512[2];
  bias512[0] = (float*)carve(512 * 4);
  bias512[1] = (float*)carve(512 * 4);

  // ---- pack weights into WMMA fragment order (cheap, done every call) ----
  for (int l = 0; l < 2; l++) {
    for (int w = 0; w < 4; w++)
      pack_weight_kernel<<<(128 * 128 + 255) / 256, 256, 0, stream>>>(
          LW[l][w], packL[l], 128, 128, 32, w * 8);
    pack_bias4_kernel<<<2, 256, 0, stream>>>(LB[l][0], LB[l][1], LB[l][2], LB[l][3], bias512[l]);
  }
  pack_weight_kernel<<<(768 * 128 + 255) / 256, 256, 0, stream>>>(Wc, packWc, 768, 128, 8, 0);
  pack_weight_kernel<<<(512 * 128 + 255) / 256, 256, 0, stream>>>(W1, packW1, 512, 128, 8, 0);

  init_z_kernel<<<(MPAD * HID) / 256, 256, 0, stream>>>(node_emb, zbf);

  // ---- 2 transformer layers ----
  for (int l = 0; l < 2; l++) {
    gemm_wmma_kernel<<<(MPAD / 256) * (512 / 64), 256, 0, stream>>>(
        zbf, 0, HID, MPAD, packL[l], bias512[l], nullptr, qkvs, nullptr, HID, 512, 0);
    init_attn_kernel<<<(MPAD * HID) / 256, 256, 0, stream>>>(agg, mbuf, denom);
    attn_scores_kernel<<<(NE + 7) / 8, 256, 0, stream>>>(edge_index, qkvs, sc, mbuf);
    attn_accum_kernel<<<(NE + 7) / 8, 256, 0, stream>>>(edge_index, qkvs, sc, mbuf, denom, agg);
    attn_finalize_kernel<<<(MPAD * HID) / 256, 256, 0, stream>>>(agg, denom, qkvs, zbf);
  }

  // ---- chemistry projection (A read as f32 with row clamp + smiles mask) ----
  gemm_wmma_kernel<<<(MPAD / 256) * (128 / 64), 256, 0, stream>>>(
      chemistry, 1, CD, N_NODES, packWc, bc, smiles_mask, nullptr, cbf, CD, 128, 0);

  // ---- decoder for pos / neg edge sets ----
  const int* evsets[2] = {pos_edge, neg_edge};
  for (int s = 0; s < 2; s++) {
    decode_gather_kernel<<<(MPAD * 64) / 256, 256, 0, stream>>>(evsets[s], zbf, cbf, adec);
    gemm_wmma_kernel<<<(MPAD / 256) * (128 / 64), 256, 0, stream>>>(
        adec, 0, 512, MPAD, packW1, b1, nullptr, nullptr, hidden, 512, 128, 1);
    decode_out_kernel<<<(EVN + 7) / 8, 256, 0, stream>>>(
        hidden, W2, b2, (float*)d_out + (size_t)s * EVN);
  }
}
